// EfficientKANLayer_6828998001414
// MI455X (gfx1250) — compile-verified, matrix operand order checked by reading, not run on hardware
//
#include <hip/hip_runtime.h>
#include <hip/hip_bf16.h>

#define IN_F   1024
#define OUT_F  256
#define NB     8
#define KSPL   (IN_F * NB)       // 8192
#define KCAT   (IN_F + KSPL)     // 9216 combined K (base ++ spline)
#define BM     64
#define BN     256
#define BK     32
#define NTHREADS 256
#define LDS_STRIDE 40            // 32 k-values + 8 pad (80 B rows: 16B aligned, conflict-free)
#define BASE_STEPS (IN_F / BK)   // 32
#define NSTEPS (KCAT / BK)       // 288

typedef __attribute__((ext_vector_type(16))) __bf16 bf16x16;
typedef __attribute__((ext_vector_type(8)))  float  f32x8;

__device__ __forceinline__ float silu_f(float v) {
    return v * __builtin_amdgcn_rcpf(1.0f + __expf(-v));
}

__device__ __forceinline__ unsigned pack_bf16x2(float a, float b) {
    union { __bf16 h[2]; unsigned u; } p;
    p.h[0] = (__bf16)a;
    p.h[1] = (__bf16)b;
    return p.u;
}

// ---- A tile, base phase: silu(x), inline loads (streaming, 32 steps only) ----
__device__ __forceinline__ void stage_A_base(int step, int t, int row0,
                                             const float* __restrict__ x, __bf16* aT) {
    const int k0 = step * BK;
#pragma unroll
    for (int i = 0; i < 2; ++i) {
        int v = t + i * NTHREADS;
        int r = v >> 3, q = v & 7;
        float4 xv = *(const float4*)(x + (size_t)(row0 + r) * IN_F + k0 + q * 4);
        uint2 pk;
        pk.x = pack_bf16x2(silu_f(xv.x), silu_f(xv.y));
        pk.y = pack_bf16x2(silu_f(xv.z), silu_f(xv.w));
        *(uint2*)(aT + r * LDS_STRIDE + q * 4) = pk;
    }
}

// ---- x prefetch for spline staging of `step` (consumed one iteration later) ----
__device__ __forceinline__ float load_x_spline(int step, int t, int row0,
                                               const float* __restrict__ x) {
    const int f = (step - BASE_STEPS) * 4 + (t & 3);
    return x[(size_t)(row0 + (t >> 2)) * IN_F + f];
}

// ---- A tile, spline phase: cubic B-spline basis from pre-loaded x ----
__device__ __forceinline__ void stage_A_spline(int t, float xv, float g0, float ginv,
                                               __bf16* aT) {
    int r = t >> 2, fc = t & 3;
    float tt = (xv - g0) * ginv;             // knot units; basis_j = N(tt - j)
    float fl = floorf(tt);
    int   ii = (int)fl;
    float u  = tt - fl;
    float u2 = u * u, u3 = u2 * u;
    float om = 1.0f - u;
    float w3 = u3 * (1.0f / 6.0f);                       // B_{ii}
    float w0 = om * om * om * (1.0f / 6.0f);             // B_{ii-3}
    float w1 = (1.0f / 6.0f) *
               (fmaf(u2, fmaf(-3.0f, u, 3.0f), fmaf(3.0f, u, 1.0f))); // B_{ii-1}
    float w2 = 1.0f - w0 - w1 - w3;                      // B_{ii-2}
    float w[4] = { w0, w2, w1, w3 };   // slots j0..j0+3 = B_{ii-3..ii}
    __bf16* rowp = aT + r * LDS_STRIDE + fc * 8;
    uint4 z = { 0u, 0u, 0u, 0u };
    *(uint4*)rowp = z;                 // zero all 8 slots (one b128 store)
    int j0 = ii - 3;
    const int padOff = 32 - fc * 8;    // row's pad slot (never read by fragments)
#pragma unroll
    for (int w_i = 0; w_i < 4; ++w_i) {
        int j = j0 + w_i;
        int off = ((unsigned)j < 8u) ? j : padOff;   // cndmask, no exec churn
        rowp[off] = (__bf16)w[w_i];
    }
}

// ---- B tile: 256 n-rows x 32 k ----
template<bool USE_WS>
__device__ __forceinline__ void stage_B(int step, int t,
                                        const float* __restrict__ fsrc, size_t fstride, int fk,
                                        const __bf16* __restrict__ wcat, __bf16* bT) {
    if constexpr (USE_WS) {
        const unsigned kb = (unsigned)step * (BK * 2u);
#pragma unroll
        for (int i = 0; i < 4; ++i) {
            int v = t + i * NTHREADS;                            // 1024 x 16B chunks
            int n = v >> 2, q = v & 3;
            unsigned goff = (unsigned)n * (KCAT * 2u) + kb + (unsigned)q * 16u;
            unsigned loff = (unsigned)(uintptr_t)(bT + n * LDS_STRIDE + q * 8);
            unsigned long long base = (unsigned long long)(uintptr_t)wcat;
            asm volatile("global_load_async_to_lds_b128 %0, %1, %2"
                         :: "v"(loff), "v"(goff), "s"(base) : "memory");
        }
    } else {
#pragma unroll
        for (int i = 0; i < 8; ++i) {
            int v = t + i * NTHREADS;
            int n = v >> 3, q = v & 7;
            float4 wv = *(const float4*)(fsrc + (size_t)n * fstride + fk + q * 4);
            uint2 pk;
            pk.x = pack_bf16x2(wv.x, wv.y);
            pk.y = pack_bf16x2(wv.z, wv.w);
            *(uint2*)(bT + n * LDS_STRIDE + q * 4) = pk;
        }
    }
}

// ---- fragment loads + 8 WMMAs for one K-step ----
__device__ __forceinline__ void mma_step(const __bf16* aBuf, const __bf16* bBuf,
                                         int wm, int wn, int laneRow, int kh,
                                         f32x8 (&acc)[2][4]) {
    union Frag { bf16x16 v; uint4 q[2]; };
    Frag a[2], b[4];
#pragma unroll
    for (int mt = 0; mt < 2; ++mt) {
        const __bf16* p = aBuf + (wm * 32 + mt * 16 + laneRow) * LDS_STRIDE + kh;
        a[mt].q[0] = *(const uint4*)p;
        a[mt].q[1] = *(const uint4*)(p + 16);
    }
#pragma unroll
    for (int nt = 0; nt < 4; ++nt) {
        const __bf16* p = bBuf + (wn * 64 + nt * 16 + laneRow) * LDS_STRIDE + kh;
        b[nt].q[0] = *(const uint4*)p;
        b[nt].q[1] = *(const uint4*)(p + 16);
    }
#pragma unroll
    for (int mt = 0; mt < 2; ++mt)
#pragma unroll
        for (int nt = 0; nt < 4; ++nt)
            acc[mt][nt] = __builtin_amdgcn_wmma_f32_16x16x32_bf16(
                false, a[mt].v, false, b[nt].v,
                (short)0, acc[mt][nt], false, false);
}

// ---------------- fused KAN forward ----------------
template<bool USE_WS>
__global__ __launch_bounds__(NTHREADS)
void kan_fused_wmma(const float* __restrict__ x,
                    const float* __restrict__ bw,
                    const float* __restrict__ sw,
                    const float* __restrict__ grid,
                    const __bf16* __restrict__ wcat,
                    float* __restrict__ out,
                    int M)
{
    __shared__ __align__(16) __bf16 ldsA[2][BM * LDS_STRIDE];   // 2 x  5 KB
    __shared__ __align__(16) __bf16 ldsB[2][BN * LDS_STRIDE];   // 2 x 20 KB

    const int t       = threadIdx.x;
    const int lane    = t & 31;
    const int wave    = t >> 5;
    const int wm      = wave >> 2;          // 0..1 (M)
    const int wn      = wave & 3;           // 0..3 (N)
    const int laneRow = lane & 15;
    const int kh      = (lane >> 4) * 8;    // WMMA 16-bit operand lane/K layout

    const int row0 = blockIdx.x * BM;

    // grid is a broadcast row (make_grid): g0 and 1/h are kernel-wide constants.
    const float g0   = grid[0];
    const float ginv = __builtin_amdgcn_rcpf(grid[1] - g0);

    const f32x8 zero = {0.f,0.f,0.f,0.f,0.f,0.f,0.f,0.f};
    f32x8 acc[2][4];
#pragma unroll
    for (int i = 0; i < 2; ++i)
#pragma unroll
        for (int j = 0; j < 4; ++j) acc[i][j] = zero;

#define SYNC_STEP()                                                     \
    do {                                                                \
        if constexpr (USE_WS)                                           \
            asm volatile("s_wait_asynccnt 0x0" ::: "memory");           \
        __syncthreads();                                                \
    } while (0)

    // prologue: stage step 0
    stage_A_base(0, t, row0, x, ldsA[0]);
    stage_B<USE_WS>(0, t, bw, IN_F, 0, wcat, ldsB[0]);

    // ---- phase 1 loop: s = 0..30 (stage s+1, all base steps) ----
    for (int s = 0; s < BASE_STEPS - 1; ++s) {
        SYNC_STEP();
        stage_A_base(s + 1, t, row0, x, ldsA[(s + 1) & 1]);
        stage_B<USE_WS>(s + 1, t, bw, IN_F, (s + 1) * BK, wcat, ldsB[(s + 1) & 1]);
        mma_step(ldsA[s & 1], ldsB[s & 1], wm, wn, laneRow, kh, acc);
    }

    // ---- boundary s = 31: stage first spline step (32) ----
    SYNC_STEP();
    {
        float xb = load_x_spline(BASE_STEPS, t, row0, x);       // single inline stall, once
        stage_A_spline(t, xb, g0, ginv, ldsA[0]);               // 32 & 1 == 0
        stage_B<USE_WS>(BASE_STEPS, t, sw, KSPL, 0, wcat, ldsB[0]);
        mma_step(ldsA[1], ldsB[1], wm, wn, laneRow, kh, acc);   // s = 31 parity
    }

    // ---- phase 2 steady loop: s = 32..286 (branch-free body) ----
    float xs = load_x_spline(BASE_STEPS + 1, t, row0, x);       // prefetch for stage 33
    for (int s = BASE_STEPS; s < NSTEPS - 1; ++s) {
        SYNC_STEP();
        int pre = (s + 2 < NSTEPS) ? s + 2 : NSTEPS - 1;        // clamp: safe redundant load
        float xn = load_x_spline(pre, t, row0, x);              // issue early
        stage_A_spline(t, xs, g0, ginv, ldsA[(s + 1) & 1]);
        stage_B<USE_WS>(s + 1, t, sw, KSPL, (s + 1 - BASE_STEPS) * BK, wcat,
                        ldsB[(s + 1) & 1]);
        xs = xn;
        mma_step(ldsA[s & 1], ldsB[s & 1], wm, wn, laneRow, kh, acc);
    }

    // ---- final iteration s = 287 ----
    SYNC_STEP();
    mma_step(ldsA[(NSTEPS - 1) & 1], ldsB[(NSTEPS - 1) & 1], wm, wn, laneRow, kh, acc);

#undef SYNC_STEP

    // writeback (C/D layout: lanes<16 -> M=r, lanes>=16 -> M=r+8, N=lane%16)
    const int rowAdd = (lane >> 4) * 8;
#pragma unroll
    for (int mt = 0; mt < 2; ++mt)
#pragma unroll
        for (int nt = 0; nt < 4; ++nt)
#pragma unroll
            for (int r = 0; r < 8; ++r) {
                int row = row0 + wm * 32 + mt * 16 + r + rowAdd;
                int col = wn * 64 + nt * 16 + laneRow;
                out[(size_t)row * OUT_F + col] = acc[mt][nt][r];
            }
}

// ---------------- one-shot weight conversion (f32 -> concatenated bf16) ----------------
__global__ __launch_bounds__(NTHREADS)
void convert_weights(const float* __restrict__ bw,
                     const float* __restrict__ sw,
                     __bf16* __restrict__ wcat)
{
    int idx = blockIdx.x * blockDim.x + threadIdx.x;   // one 4-element chunk
    int n   = idx / (KCAT / 4);
    int k4  = (idx % (KCAT / 4)) * 4;
    float4 v = (k4 < IN_F)
        ? *(const float4*)(bw + (size_t)n * IN_F + k4)
        : *(const float4*)(sw + (size_t)n * KSPL + (k4 - IN_F));
    uint2 pk;
    pk.x = pack_bf16x2(v.x, v.y);
    pk.y = pack_bf16x2(v.z, v.w);
    *(uint2*)(wcat + (size_t)n * KCAT + k4) = pk;
}

extern "C" void kernel_launch(void* const* d_in, const int* in_sizes, int n_in,
                              void* d_out, int out_size, void* d_ws, size_t ws_size,
                              hipStream_t stream) {
    const float* x  = (const float*)d_in[0];
    const float* bw = (const float*)d_in[1];
    const float* sw = (const float*)d_in[2];
    const float* gr = (const float*)d_in[3];
    float* out = (float*)d_out;
    (void)n_in; (void)out_size;

    int M = in_sizes[0] / IN_F;          // 16384
    dim3 grid(M / BM);                   // 256 workgroups

    const size_t wcat_bytes = (size_t)OUT_F * KCAT * sizeof(__bf16);  // 4.72 MB
    if (ws_size >= wcat_bytes) {
        __bf16* wcat = (__bf16*)d_ws;
        int chunks = OUT_F * (KCAT / 4);
        convert_weights<<<chunks / NTHREADS, NTHREADS, 0, stream>>>(bw, sw, wcat);
        kan_fused_wmma<true><<<grid, NTHREADS, 0, stream>>>(x, bw, sw, gr, wcat, out, M);
    } else {
        kan_fused_wmma<false><<<grid, NTHREADS, 0, stream>>>(x, bw, sw, gr, nullptr, out, M);
    }
}
